// DialogueRNNCell_1494648619060
// MI455X (gfx1250) — matrix operand order repaired and checked by reading, not compiled, not run on hardware
//
#include <hip/hip_runtime.h>
#include <hip/hip_bf16.h>

// ---------------- problem constants ----------------
#define B_   2048
#define T_   128
#define P_   8
#define DM   512
#define DG   150
#define DP   150
#define DE   100

// padded dims (K -> mult of 32, N -> mult of 32 for 32-wide N blocking)
#define KX   672   // D_m + D_p = 662 -> 672
#define NG   480   // 3*150 = 450 -> 480
#define KH   160   // 150 -> 160
#define NE   128   // 100 -> 128
#define KEH  128   // 100 -> 128

typedef __attribute__((ext_vector_type(16))) __bf16 v16bf;
typedef __attribute__((ext_vector_type(8)))  float  v8f;

union FragBF { unsigned int u[8]; v16bf v; };

__device__ __forceinline__ unsigned short f2bf(float f) {
  unsigned int u = __float_as_uint(f);
  u += 0x7FFFu + ((u >> 16) & 1u);   // round-to-nearest-even
  return (unsigned short)(u >> 16);
}
__device__ __forceinline__ float sigm(float x) { return 1.0f / (1.0f + __expf(-x)); }

// ---------------- cast + zero-pad f32 -> bf16 ----------------
__global__ void k_cast_pad(const float* __restrict__ src, unsigned short* __restrict__ dst,
                           int srows, int scols, int drows, int dcols) {
  int idx = blockIdx.x * blockDim.x + threadIdx.x;
  if (idx >= drows * dcols) return;
  int r = idx / dcols, c = idx % dcols;
  float v = (r < srows && c < scols) ? src[(size_t)r * scols + c] : 0.0f;
  dst[idx] = f2bf(v);
}

// ---------------- speaker argmax ----------------
__global__ void k_qmidx(const float* __restrict__ qmask, int* __restrict__ qidx) {
  int b = blockIdx.x * blockDim.x + threadIdx.x;
  if (b >= B_) return;
  float best = qmask[b * P_]; int bi = 0;
  for (int p = 1; p < P_; ++p) { float v = qmask[b * P_ + p]; if (v > best) { best = v; bi = p; } }
  qidx[b] = bi;
}

// gather q0_sel (f32 + bf16) and place into X1 columns [DM, DM+KH)
__global__ void k_gather(const float* __restrict__ q0, const int* __restrict__ qidx,
                         float* __restrict__ q0sel_f, unsigned short* __restrict__ q0sel_bf,
                         unsigned short* __restrict__ X1) {
  int idx = blockIdx.x * blockDim.x + threadIdx.x;
  if (idx >= B_ * KH) return;
  int b = idx / KH, h = idx % KH;
  if (h < DP) {
    float v = q0[((size_t)b * P_ + qidx[b]) * DP + h];
    q0sel_f[(size_t)b * DP + h] = v;
    unsigned short bv = f2bf(v);
    q0sel_bf[(size_t)b * KH + h] = bv;
    X1[(size_t)b * KX + DM + h] = bv;
  } else {
    q0sel_bf[(size_t)b * KH + h] = 0;
    X1[(size_t)b * KX + DM + h] = 0;
  }
}

// ---------------- attention: scale -> softmax(T) -> context; one block per b ----------------
__global__ void k_attention(const float* __restrict__ g_hist, const float* __restrict__ att_w,
                            float* __restrict__ out_alpha, unsigned short* __restrict__ X2) {
  __shared__ float s_alpha[T_];
  __shared__ float s_red[T_];
  int b = blockIdx.x;
  int t = threadIdx.x;          // 0..127
  const float* gp = g_hist + ((size_t)t * B_ + b) * DG;
  float acc = 0.0f;
  for (int d = 0; d < DG; ++d) acc += gp[d] * att_w[d];
  s_red[t] = acc;
  __syncthreads();
  for (int s = T_ / 2; s > 0; s >>= 1) { if (t < s) s_red[t] = fmaxf(s_red[t], s_red[t + s]); __syncthreads(); }
  float mx = s_red[0];
  __syncthreads();
  float ex = __expf(acc - mx);
  s_red[t] = ex;
  __syncthreads();
  for (int s = T_ / 2; s > 0; s >>= 1) { if (t < s) s_red[t] += s_red[t + s]; __syncthreads(); }
  float alpha = ex / s_red[0];
  s_alpha[t] = alpha;
  __syncthreads();
  out_alpha[(size_t)b * T_ + t] = alpha;          // alpha.T[:,None,:] -> flat b*T + t
  for (int d = t; d < DG; d += T_) {
    float c = 0.0f;
    for (int tt = 0; tt < T_; ++tt) c += s_alpha[tt] * g_hist[((size_t)tt * B_ + b) * DG + d];
    X2[(size_t)b * KX + DM + d] = f2bf(c);
  }
}

// ---------------- WMMA GEMM: C(MxN) = A(MxK) @ W(NxK)^T, bf16 in, f32 accum ----------------
// Block = 128 threads (4 waves). Block tile: M=128 (wave w -> rows w*32..w*32+31), N=32.
// W panel (32 x K) staged once per block into LDS via global_load_async_to_lds_b128
// (ASYNCcnt), then each wave does a 32x32 register tile: 4 WMMAs per 32-K step.
// Requires: M % 128 == 0, N % 32 == 0, K % 32 == 0. Dynamic LDS = 64*K bytes.
__global__ void k_gemm_bf16(const unsigned short* __restrict__ A,
                            const unsigned short* __restrict__ W,
                            float* __restrict__ C,
                            int M, int N, int K) {
  extern __shared__ unsigned short sW[];    // 32 x K bf16 panel
  int tid  = threadIdx.x;
  int lane = tid & 31;
  int wave = tid >> 5;                      // 0..3
  int nt0  = blockIdx.y * 32;
  int mt0  = blockIdx.x * 128 + wave * 32;

  // ---- async-stage W panel rows [nt0, nt0+32) x K into LDS (coalesced 16B chunks) ----
  {
    const unsigned short* wPanel = W + (size_t)nt0 * K;
    unsigned ldsBase = (unsigned)(size_t)&sW[0];
    int iters = K >> 5;                     // (32*K elems)/(128 thr * 8 elems)
    for (int i = 0; i < iters; ++i) {
      int elem = (i * 128 + tid) * 8;       // ushort index of 16B chunk
      unsigned lofs = ldsBase + (unsigned)elem * 2u;
      const unsigned short* g = wPanel + elem;
      asm volatile("global_load_async_to_lds_b128 %0, %1, off"
                   :: "v"(lofs), "v"(g) : "memory");
    }
    asm volatile("s_wait_asynccnt 0x0" ::: "memory");
  }
  __syncthreads();

  int row = lane & 15;
  int hi  = lane >> 4;
  const unsigned short* aRow0 = A + (size_t)(mt0 + row) * K;
  const unsigned short* aRow1 = A + (size_t)(mt0 + 16 + row) * K;
  const unsigned short* w0Row = &sW[(size_t)row * K];
  const unsigned short* w1Row = &sW[(size_t)(16 + row) * K];

  v8f acc00 = {}, acc01 = {}, acc10 = {}, acc11 = {};
  for (int k0 = 0; k0 < K; k0 += 32) {
    if (k0 + 32 < K) {
      __builtin_prefetch(aRow0 + k0 + 32, 0, 3);   // global_prefetch_b8
      __builtin_prefetch(aRow1 + k0 + 32, 0, 3);
    }
    FragBF a0, a1, w0, w1;
#pragma unroll
    for (int j = 0; j < 8; ++j) {
      // A (16x32, MxK): lanes 0-15 hold K {0..7,16..23}; lanes 16-31 hold K {8..15,24..31}
      int kkA = (j < 4) ? (hi * 8 + 2 * j) : (16 + hi * 8 + 2 * (j - 4));
      // B (32x16, KxN): lanes 0-15 hold K {0..7,16..23}; lanes 16-31 hold K {8..15,24..31}
      int kkB = kkA;
      a0.u[j] = *(const unsigned int*)(aRow0 + k0 + kkA);
      a1.u[j] = *(const unsigned int*)(aRow1 + k0 + kkA);
      w0.u[j] = *(const unsigned int*)(w0Row + k0 + kkB);
      w1.u[j] = *(const unsigned int*)(w1Row + k0 + kkB);
    }
    acc00 = __builtin_amdgcn_wmma_f32_16x16x32_bf16(false, a0.v, false, w0.v, (short)0, acc00, false, false);
    acc01 = __builtin_amdgcn_wmma_f32_16x16x32_bf16(false, a0.v, false, w1.v, (short)0, acc01, false, false);
    acc10 = __builtin_amdgcn_wmma_f32_16x16x32_bf16(false, a1.v, false, w0.v, (short)0, acc10, false, false);
    acc11 = __builtin_amdgcn_wmma_f32_16x16x32_bf16(false, a1.v, false, w1.v, (short)0, acc11, false, false);
  }
  // C/D layout: VGPR i -> (M = i + 8*hi, N = lane&15)
  int col0 = nt0 + (lane & 15);
  int rb0  = mt0 + hi * 8;
#pragma unroll
  for (int i = 0; i < 8; ++i) {
    C[(size_t)(rb0 + i)      * N + col0]      = acc00[i];
    C[(size_t)(rb0 + i)      * N + col0 + 16] = acc01[i];
    C[(size_t)(rb0 + 16 + i) * N + col0]      = acc10[i];
    C[(size_t)(rb0 + 16 + i) * N + col0 + 16] = acc11[i];
  }
}

// ---------------- GRU gate fusion for g_ ----------------
__global__ void k_gru_g(const float* __restrict__ gi, const float* __restrict__ gh,
                        const float* __restrict__ bih, const float* __restrict__ bhh,
                        const float* __restrict__ hprev, float* __restrict__ out_g) {
  int idx = blockIdx.x * blockDim.x + threadIdx.x;
  if (idx >= B_ * DG) return;
  int b = idx / DG, h = idx % DG;
  const float* gib = gi + (size_t)b * NG;
  const float* ghb = gh + (size_t)b * NG;
  float ir = gib[h]          + bih[h];
  float iz = gib[DG + h]     + bih[DG + h];
  float in = gib[2 * DG + h] + bih[2 * DG + h];
  float hr = ghb[h]          + bhh[h];
  float hz = ghb[DG + h]     + bhh[DG + h];
  float hn = ghb[2 * DG + h] + bhh[2 * DG + h];
  float r = sigm(ir + hr), z = sigm(iz + hz);
  float n = tanhf(in + r * hn);
  out_g[idx] = (1.0f - z) * n + z * hprev[(size_t)b * DG + h];
}

// ---------------- q_ = q0 everywhere (selected rows overwritten next) ----------------
__global__ void k_copy_q(const float* __restrict__ q0, float* __restrict__ out_q) {
  int idx = blockIdx.x * blockDim.x + threadIdx.x;
  if (idx < B_ * P_ * DP) out_q[idx] = q0[idx];
}

// ---------------- GRU gate fusion for selected-speaker party state ----------------
__global__ void k_gru_p(const float* __restrict__ gi, const float* __restrict__ gh,
                        const float* __restrict__ bih, const float* __restrict__ bhh,
                        const float* __restrict__ hprev, const int* __restrict__ qidx,
                        float* __restrict__ out_q, unsigned short* __restrict__ qs_bf) {
  int idx = blockIdx.x * blockDim.x + threadIdx.x;
  if (idx >= B_ * KH) return;
  int b = idx / KH, h = idx % KH;
  if (h >= DP) { qs_bf[(size_t)b * KH + h] = 0; return; }
  const float* gib = gi + (size_t)b * NG;
  const float* ghb = gh + (size_t)b * NG;
  float ir = gib[h]          + bih[h];
  float iz = gib[DP + h]     + bih[DP + h];
  float in = gib[2 * DP + h] + bih[2 * DP + h];
  float hr = ghb[h]          + bhh[h];
  float hz = ghb[DP + h]     + bhh[DP + h];
  float hn = ghb[2 * DP + h] + bhh[2 * DP + h];
  float r = sigm(ir + hr), z = sigm(iz + hz);
  float n = tanhf(in + r * hn);
  float qs = (1.0f - z) * n + z * hprev[(size_t)b * DP + h];
  out_q[((size_t)b * P_ + qidx[b]) * DP + h] = qs;
  qs_bf[(size_t)b * KH + h] = f2bf(qs);
}

// ---------------- emotion output ----------------
__global__ void k_emotion(const float* __restrict__ ei, const float* __restrict__ eh,
                          const float* __restrict__ bih, const float* __restrict__ bhh,
                          float* __restrict__ out_e) {
  int idx = blockIdx.x * blockDim.x + threadIdx.x;
  if (idx >= B_ * DE) return;
  int b = idx / DE, h = idx % DE;
  out_e[idx] = tanhf(ei[(size_t)b * NE + h] + bih[h] + eh[(size_t)b * NE + h] + bhh[h]);
}

// =====================================================================
extern "C" void kernel_launch(void* const* d_in, const int* in_sizes, int n_in,
                              void* d_out, int out_size, void* d_ws, size_t ws_size,
                              hipStream_t stream) {
  const float* U      = (const float*)d_in[0];
  const float* qmask  = (const float*)d_in[1];
  const float* g_hist = (const float*)d_in[2];
  const float* q0     = (const float*)d_in[3];
  const float* e0     = (const float*)d_in[4];
  const float* g_Wih  = (const float*)d_in[5];
  const float* g_Whh  = (const float*)d_in[6];
  const float* g_bih  = (const float*)d_in[7];
  const float* g_bhh  = (const float*)d_in[8];
  const float* p_Wih  = (const float*)d_in[9];
  const float* p_Whh  = (const float*)d_in[10];
  const float* p_bih  = (const float*)d_in[11];
  const float* p_bhh  = (const float*)d_in[12];
  const float* e_Wih  = (const float*)d_in[13];
  const float* e_Whh  = (const float*)d_in[14];
  const float* e_bih  = (const float*)d_in[15];
  const float* e_bhh  = (const float*)d_in[16];
  const float* att_w  = (const float*)d_in[17];

  float* out   = (float*)d_out;
  float* out_g = out;
  float* out_q = out_g + (size_t)B_ * DG;
  float* out_e = out_q + (size_t)B_ * P_ * DP;
  float* out_a = out_e + (size_t)B_ * DE;

  char* ws = (char*)d_ws;
  size_t off = 0;
  auto alloc = [&](size_t bytes) -> void* {
    void* p = ws + off;
    off = (off + bytes + 255) & ~(size_t)255;
    return p;
  };

  int*            qidx     = (int*)            alloc((size_t)B_ * 4);
  unsigned short* X1       = (unsigned short*) alloc((size_t)B_ * KX * 2);   // [U | q0_sel]
  unsigned short* X2       = (unsigned short*) alloc((size_t)B_ * KX * 2);   // [U | c_]
  unsigned short* Hg       = (unsigned short*) alloc((size_t)B_ * KH * 2);   // g_hist[-1]
  float*          q0sel_f  = (float*)          alloc((size_t)B_ * DP * 4);
  unsigned short* q0sel_bf = (unsigned short*) alloc((size_t)B_ * KH * 2);
  unsigned short* e0_bf    = (unsigned short*) alloc((size_t)B_ * KEH * 2);
  unsigned short* qs_bf    = (unsigned short*) alloc((size_t)B_ * KH * 2);
  unsigned short* Wgih     = (unsigned short*) alloc((size_t)NG * KX * 2);
  unsigned short* Wghh     = (unsigned short*) alloc((size_t)NG * KH * 2);
  unsigned short* Wpih     = (unsigned short*) alloc((size_t)NG * KX * 2);
  unsigned short* Wphh     = (unsigned short*) alloc((size_t)NG * KH * 2);
  unsigned short* Weih     = (unsigned short*) alloc((size_t)NE * KH * 2);
  unsigned short* Wehh     = (unsigned short*) alloc((size_t)NE * KEH * 2);
  float*          gi_g     = (float*)          alloc((size_t)B_ * NG * 4);
  float*          gh_g     = (float*)          alloc((size_t)B_ * NG * 4);
  float*          gi_p     = (float*)          alloc((size_t)B_ * NG * 4);
  float*          gh_p     = (float*)          alloc((size_t)B_ * NG * 4);
  float*          ei       = (float*)          alloc((size_t)B_ * NE * 4);
  float*          eh       = (float*)          alloc((size_t)B_ * NE * 4);

  auto cast_pad = [&](const float* src, unsigned short* dst, int sr, int sc, int dr, int dc) {
    int total = dr * dc;
    k_cast_pad<<<(total + 255) / 256, 256, 0, stream>>>(src, dst, sr, sc, dr, dc);
  };
  auto gemm = [&](const unsigned short* A, const unsigned short* W, float* C, int M, int N, int K) {
    dim3 grid(M / 128, N / 32);
    size_t ldsBytes = (size_t)64 * K;       // 32 rows x K bf16
    k_gemm_bf16<<<grid, 128, ldsBytes, stream>>>(A, W, C, M, N, K);
  };

  const float* g_last = g_hist + (size_t)(T_ - 1) * B_ * DG;

  // ---- prep ----
  k_qmidx<<<(B_ + 255) / 256, 256, 0, stream>>>(qmask, qidx);
  cast_pad(U,      X1,  B_, DM,  B_, KX);   // zeros cols [DM,KX)
  cast_pad(U,      X2,  B_, DM,  B_, KX);
  cast_pad(g_last, Hg,  B_, DG,  B_, KH);
  cast_pad(g_Wih,  Wgih, 3 * DG, DM + DP, NG, KX);
  cast_pad(g_Whh,  Wghh, 3 * DG, DG,      NG, KH);
  cast_pad(p_Wih,  Wpih, 3 * DP, DM + DG, NG, KX);
  cast_pad(p_Whh,  Wphh, 3 * DP, DP,      NG, KH);
  cast_pad(e_Wih,  Weih, DE, DP, NE, KH);
  cast_pad(e_Whh,  Wehh, DE, DE, NE, KEH);
  cast_pad(e0,     e0_bf, B_, DE, B_, KEH);
  k_gather<<<(B_ * KH + 255) / 256, 256, 0, stream>>>(q0, qidx, q0sel_f, q0sel_bf, X1);

  // ---- attention (alpha output + context into X2) ----
  k_attention<<<B_, T_, 0, stream>>>(g_hist, att_w, out_a, X2);

  // ---- WMMA GEMMs ----
  gemm(X1,       Wgih, gi_g, B_, NG, KX);   // g-cell input gates
  gemm(Hg,       Wghh, gh_g, B_, NG, KH);   // g-cell hidden gates
  gemm(X2,       Wpih, gi_p, B_, NG, KX);   // p-cell input gates
  gemm(q0sel_bf, Wphh, gh_p, B_, NG, KH);   // p-cell hidden gates (selected speaker only)

  // ---- fused GRU elementwise ----
  k_gru_g<<<(B_ * DG + 255) / 256, 256, 0, stream>>>(gi_g, gh_g, g_bih, g_bhh, g_last, out_g);
  k_copy_q<<<(B_ * P_ * DP + 255) / 256, 256, 0, stream>>>(q0, out_q);
  k_gru_p<<<(B_ * KH + 255) / 256, 256, 0, stream>>>(gi_p, gh_p, p_bih, p_bhh,
                                                     q0sel_f, qidx, out_q, qs_bf);

  // ---- emotion GEMMs + tanh ----
  gemm(qs_bf, Weih, ei, B_, NE, KH);
  gemm(e0_bf, Wehh, eh, B_, NE, KEH);
  k_emotion<<<(B_ * DE + 255) / 256, 256, 0, stream>>>(ei, eh, e_bih, e_bhh, out_e);
}
